// DramaWorldModelComponent_61452392071570
// MI455X (gfx1250) — matrix operand order repaired and checked by reading, not compiled, not run on hardware
//
#include <hip/hip_runtime.h>
#include <hip/hip_bf16.h>
#include <stdint.h>

// ---- model dims ----
constexpr int Bsz  = 4;
constexpr int T    = 512;
constexpr int STOCH= 1024;
constexpr int DM   = 512;          // d_model
constexpr int NL   = 4;
constexpr int DST  = 16;           // d_state
constexpr int DI   = 1024;         // d_inner
constexpr int DTR  = 32;           // dt_rank
constexpr int M    = Bsz * T;      // 2048 rows in every GEMM

constexpr int AROW = 40;           // padded LDS row stride (elems) for A strips

typedef __attribute__((ext_vector_type(8)))  float   v8f;
typedef __attribute__((ext_vector_type(8)))  __bf16  v8bf;
typedef __attribute__((ext_vector_type(16))) __bf16  v16bf;
typedef __attribute__((ext_vector_type(4)))  int     v4i;

typedef __attribute__((address_space(1))) v4i gv4i;   // global int4
typedef __attribute__((address_space(3))) v4i lv4i;   // LDS int4

#if __has_builtin(__builtin_amdgcn_global_load_async_to_lds_b128)
#define HAVE_ASYNC_LDS 1
#else
#define HAVE_ASYNC_LDS 0
#endif

#if HAVE_ASYNC_LDS
# if __has_builtin(__builtin_amdgcn_s_wait_asynccnt)
#  define ASYNC_WAIT(n) __builtin_amdgcn_s_wait_asynccnt(n)
# else
#  define ASYNC_WAIT(n) asm volatile("s_wait_asynccnt " #n ::: "memory")
# endif
#else
# define ASYNC_WAIT(n) do { } while (0)
#endif

__device__ __forceinline__ unsigned short f2bf(float f) {
  unsigned int u = __float_as_uint(f);
  u += 0x7FFFu + ((u >> 16) & 1u);          // round-to-nearest-even
  return (unsigned short)(u >> 16);
}

__device__ __forceinline__ void copy16_async(const unsigned short* g,
                                             unsigned short* l) {
#if HAVE_ASYNC_LDS
  __builtin_amdgcn_global_load_async_to_lds_b128((gv4i*)g, (lv4i*)l, 0, 0);
#else
  *(v8bf*)l = *(const v8bf*)g;
#endif
}

// stage 4KB B strip (4 n-tiles x 32 lanes x 16 elems, contiguous in global)
__device__ __forceinline__ void stage_b_strip(const unsigned short* __restrict__ g,
                                              unsigned short* l) {
  int tid = threadIdx.x;                    // 0..127
  copy16_async(g + (size_t)tid * 8,         l + (size_t)tid * 8);
  copy16_async(g + (size_t)(tid + 128) * 8, l + (size_t)(tid + 128) * 8);
}

// stage 64-row x 32-K A strip (row-major, lds row stride AROW elems)
__device__ __forceinline__ void stage_a_strip(const unsigned short* __restrict__ g,
                                              int K, unsigned short* l) {
  int tid = threadIdx.x;                    // 256 chunks of 16B
  int c0 = tid, c1 = tid + 128;
  copy16_async(g + (size_t)(c0 >> 2) * K + (c0 & 3) * 8,
               l + (size_t)(c0 >> 2) * AROW + (c0 & 3) * 8);
  copy16_async(g + (size_t)(c1 >> 2) * K + (c1 & 3) * 8,
               l + (size_t)(c1 >> 2) * AROW + (c1 & 3) * 8);
}

// one K-step: A fragment + 4 B fragments from LDS, 4 WMMAs
__device__ __forceinline__ void gemm_step(const unsigned short* abuf,
                                          const unsigned short* bbuf,
                                          int wave, int lane, v8f acc[4]) {
  int laneL = lane & 15;
  int hi    = lane >> 4;
  int arow  = (wave * 16 + laneL) * AROW + hi * 8;
  v8bf alo = *(const v8bf*)(abuf + arow);
  v8bf ahi = *(const v8bf*)(abuf + arow + 16);
  v16bf b0 = *(const v16bf*)(bbuf + 0 * 512 + lane * 16);
  v16bf b1 = *(const v16bf*)(bbuf + 1 * 512 + lane * 16);
  v16bf b2 = *(const v16bf*)(bbuf + 2 * 512 + lane * 16);
  v16bf b3 = *(const v16bf*)(bbuf + 3 * 512 + lane * 16);
  v16bf a  = __builtin_shufflevector(alo, ahi,
               0,1,2,3,4,5,6,7,8,9,10,11,12,13,14,15);
  acc[0] = __builtin_amdgcn_wmma_f32_16x16x32_bf16(false, a, false, b0, (short)0, acc[0], false, false);
  acc[1] = __builtin_amdgcn_wmma_f32_16x16x32_bf16(false, a, false, b1, (short)0, acc[1], false, false);
  acc[2] = __builtin_amdgcn_wmma_f32_16x16x32_bf16(false, a, false, b2, (short)0, acc[2], false, false);
  acc[3] = __builtin_amdgcn_wmma_f32_16x16x32_bf16(false, a, false, b3, (short)0, acc[3], false, false);
}

// ---------------------------------------------------------------------------
// prep kernels
// ---------------------------------------------------------------------------
__global__ void k_f32_to_bf16(const float* __restrict__ src,
                              unsigned short* __restrict__ dst, int n) {
  int i = blockIdx.x * 256 + threadIdx.x;
  if (i < n) dst[i] = f2bf(src[i]);
}

// Pack row-major KxN fp32 weight into WMMA B-fragment order (bf16)
__global__ void k_pack_b(const float* __restrict__ W,
                         unsigned short* __restrict__ P, int K, int N) {
  int idx = blockIdx.x * 256 + threadIdx.x;
  if (idx >= K * N) return;
  int j    = idx & 15;
  int l    = (idx >> 4) & 31;
  int tile = idx >> 9;
  int NT   = N >> 4;
  int nt   = tile % NT;
  int kt   = tile / NT;
  int col  = nt * 16 + (l & 15);
  int row  = kt * 32 + ((j >> 3) << 4) + ((l >> 4) << 3) + (j & 7);
  P[idx] = f2bf(W[(size_t)row * N + col]);
}

__global__ void k_aneg(const float* __restrict__ Alog,
                       float* __restrict__ An, int n) {
  int i = blockIdx.x * 256 + threadIdx.x;
  if (i < n) An[i] = -__expf(Alog[i]);
}

// ---------------------------------------------------------------------------
// bf16 WMMA GEMM:  C[M,N] (+)= A[M,K](bf16,row-major) x Bpacked[K,N]
// 4 waves/block; both A (64x32 strip) and B (32x64 strip) staged to LDS via
// async global->LDS, double-buffered; steady loop has no global loads.
// grid: (N/64, M/64), block 128.
// ---------------------------------------------------------------------------
__global__ void k_gemm_bf16(const unsigned short* __restrict__ A,
                            const unsigned short* __restrict__ Bp,
                            float* __restrict__ C,
                            int K, int N, int accum) {
  __shared__ unsigned short astage[2][64 * AROW];  // 2 x 5KB
  __shared__ unsigned short bstage[2][2048];       // 2 x 4KB

  int tid   = threadIdx.x;
  int lane  = tid & 31;
  int wave  = tid >> 5;                        // 0..3
  int mBase = blockIdx.y * 64;
  int m0    = mBase + wave * 16;
  int n0    = blockIdx.x * 64;
  int laneL = lane & 15;
  int hi    = lane >> 4;
  int NT    = N >> 4;
  int nt0   = n0 >> 4;
  int KT    = K >> 5;

  v8f acc[4];
#pragma unroll
  for (int j = 0; j < 4; ++j)
#pragma unroll
    for (int v = 0; v < 8; ++v) acc[j][v] = 0.f;

  const unsigned short* aStrip0 = A + (size_t)mBase * K;    // + kt*32 per step
  const unsigned short* bStrip0 = Bp + (size_t)nt0 * 512;   // + kt*NT*512

  // prologue: stage kt=0 (4 async issues per wave)
  stage_a_strip(aStrip0, K, astage[0]);
  stage_b_strip(bStrip0, bstage[0]);

  // steady state: branch-free, no global loads
  for (int kt = 0; kt < KT - 1; ++kt) {
    int cur = kt & 1;
    stage_a_strip(aStrip0 + (size_t)(kt + 1) * 32, K, astage[cur ^ 1]);
    stage_b_strip(bStrip0 + (size_t)(kt + 1) * NT * 512, bstage[cur ^ 1]);
    ASYNC_WAIT(4);                        // step kt's 4 issues landed
    __syncthreads();                      // all waves' portions visible
    __builtin_prefetch(bStrip0 + (size_t)(kt + 2) * NT * 512 + tid * 16, 0, 3);
    gemm_step(astage[cur], bstage[cur], wave, lane, acc);
    __syncthreads();                      // reads done before buffer reuse
  }

  // epilogue: last K-step
  ASYNC_WAIT(0);
  __syncthreads();
  gemm_step(astage[(KT - 1) & 1], bstage[(KT - 1) & 1], wave, lane, acc);

  int rBase = m0 + hi * 8;
#pragma unroll
  for (int j = 0; j < 4; ++j)
#pragma unroll
    for (int v = 0; v < 8; ++v) {
      size_t idx = (size_t)(rBase + v) * N + n0 + j * 16 + laneL;
      float val = acc[j][v];
      if (accum) val += C[idx];
      C[idx] = val;
    }
}

// ---------------------------------------------------------------------------
// embedding epilogue: x += b_embed + act_embed[actions]
// ---------------------------------------------------------------------------
__global__ void k_embed_epi(float* __restrict__ x,
                            const float* __restrict__ be,
                            const float* __restrict__ ae,
                            const int* __restrict__ act, int n) {
  int i = blockIdx.x * 256 + threadIdx.x;
  if (i >= n) return;
  int d   = i & (DM - 1);
  int row = i >> 9;                 // DM == 512
  x[i] += be[d] + ae[(size_t)act[row] * DM + d];
}

// ---------------------------------------------------------------------------
// block reduction helper (block == 256)
// ---------------------------------------------------------------------------
__device__ __forceinline__ float blockSum(float v, float* sbuf) {
  int tid = threadIdx.x;
  sbuf[tid] = v; __syncthreads();
  for (int s = 128; s > 0; s >>= 1) {
    if (tid < s) sbuf[tid] += sbuf[tid + s];
    __syncthreads();
  }
  float r = sbuf[0]; __syncthreads();
  return r;
}

// layernorm over DM=512, output bf16  (grid = M rows, block 256)
__global__ void k_ln_bf16(const float* __restrict__ x,
                          const float* __restrict__ w,
                          const float* __restrict__ b,
                          unsigned short* __restrict__ out) {
  __shared__ float sbuf[256];
  int row = blockIdx.x, tid = threadIdx.x;
  const float* xr = x + (size_t)row * DM;
  float a0 = xr[tid], a1 = xr[tid + 256];
  float m  = blockSum(a0 + a1, sbuf) * (1.f / DM);
  float d0 = a0 - m, d1 = a1 - m;
  float var = blockSum(d0 * d0 + d1 * d1, sbuf) * (1.f / DM);
  float rs = rsqrtf(var + 1e-5f);
  out[(size_t)row * DM + tid]       = f2bf(d0 * rs * w[tid] + b[tid]);
  out[(size_t)row * DM + tid + 256] = f2bf(d1 * rs * w[tid + 256] + b[tid + 256]);
}

// depthwise causal conv (K=4) + bias + SiLU.  grid (DI/256, T, B)
__global__ void k_conv_silu(const float* __restrict__ xz,
                            const float* __restrict__ cw,
                            const float* __restrict__ cb,
                            float* __restrict__ xc,
                            unsigned short* __restrict__ xcbf) {
  int d = blockIdx.x * 256 + threadIdx.x;
  int t = blockIdx.y, b = blockIdx.z;
  float acc = cb[d];
#pragma unroll
  for (int k = 0; k < 4; ++k) {
    int tt = t - 3 + k;
    if (tt >= 0)
      acc += cw[d * 4 + k] * xz[((size_t)(b * T + tt)) * (2 * DI) + d];
  }
  float s = acc / (1.f + __expf(-acc));     // SiLU
  size_t idx = ((size_t)(b * T + t)) * DI + d;
  xc[idx]   = s;
  xcbf[idx] = f2bf(s);
}

// slice dbc[:, :32] -> bf16 dt_raw
__global__ void k_dtraw(const float* __restrict__ dbc,
                        unsigned short* __restrict__ o, int n) {
  int i = blockIdx.x * 256 + threadIdx.x;
  if (i >= n) return;
  int c = i & (DTR - 1);
  int row = i >> 5;
  o[i] = f2bf(dbc[(size_t)row * 64 + c]);
}

// ---------------------------------------------------------------------------
// selective scan: one lane per channel, h[16] in registers, sequential over T.
// grid (DI/256, B), block 256
// ---------------------------------------------------------------------------
__global__ void k_scan(const float* __restrict__ xc,
                       const float* __restrict__ dtpre,
                       const float* __restrict__ dbc,
                       const float* __restrict__ xz,
                       const float* __restrict__ An,
                       const float* __restrict__ bdt,
                       const float* __restrict__ Dsk,
                       unsigned short* __restrict__ ybf) {
  int d = blockIdx.x * 256 + threadIdx.x;
  int b = blockIdx.y;
  float A[DST];
#pragma unroll
  for (int s = 0; s < DST; ++s) A[s] = An[(size_t)d * DST + s];
  float h[DST];
#pragma unroll
  for (int s = 0; s < DST; ++s) h[s] = 0.f;
  float bd = bdt[d], Dv = Dsk[d];

  for (int t = 0; t < T; ++t) {
    size_t row = (size_t)b * T + t;
    float xt = xc[row * DI + d];
    float v  = dtpre[row * DI + d] + bd;
    float dt = (v > 20.f) ? v : log1pf(__expf(v));   // softplus
    float dtx = dt * xt;
    const float* bc = dbc + row * 64 + DTR;          // B_t[16], C_t[16] (uniform)
    float y = 0.f;
#pragma unroll
    for (int s = 0; s < DST; ++s) {
      float dA = __expf(dt * A[s]);
      h[s] = dA * h[s] + dtx * bc[s];
      y   += h[s] * bc[DST + s];
    }
    float yf = y + Dv * xt;
    float z  = xz[row * (2 * DI) + DI + d];
    float sz = z / (1.f + __expf(-z));
    ybf[row * DI + d] = f2bf(yf * sz);
  }
}

// final: LN(normf) then LN(out_ln), fp32 out.  grid = M rows
__global__ void k_final_ln(const float* __restrict__ x,
                           const float* __restrict__ w1, const float* __restrict__ b1,
                           const float* __restrict__ w2, const float* __restrict__ b2,
                           float* __restrict__ out) {
  __shared__ float sbuf[256];
  int row = blockIdx.x, tid = threadIdx.x;
  const float* xr = x + (size_t)row * DM;
  float a0 = xr[tid], a1 = xr[tid + 256];
  float m  = blockSum(a0 + a1, sbuf) * (1.f / DM);
  float d0 = a0 - m, d1 = a1 - m;
  float var = blockSum(d0 * d0 + d1 * d1, sbuf) * (1.f / DM);
  float rs  = rsqrtf(var + 1e-5f);
  float y0  = d0 * rs * w1[tid] + b1[tid];
  float y1  = d1 * rs * w1[tid + 256] + b1[tid + 256];
  float m2  = blockSum(y0 + y1, sbuf) * (1.f / DM);
  float e0  = y0 - m2, e1 = y1 - m2;
  float v2  = blockSum(e0 * e0 + e1 * e1, sbuf) * (1.f / DM);
  float rs2 = rsqrtf(v2 + 1e-5f);
  out[(size_t)row * DM + tid]       = e0 * rs2 * w2[tid] + b2[tid];
  out[(size_t)row * DM + tid + 256] = e1 * rs2 * w2[tid + 256] + b2[tid + 256];
}

// mean over T.  grid (B, DM/256)
__global__ void k_mean(const float* __restrict__ ln2, float* __restrict__ out) {
  int d = blockIdx.y * 256 + threadIdx.x;
  int b = blockIdx.x;
  float s = 0.f;
  for (int t = 0; t < T; ++t)
    s += ln2[((size_t)(b * T + t)) * DM + d];
  out[(size_t)b * DM + d] = s * (1.f / T);
}

// ---------------------------------------------------------------------------
extern "C" void kernel_launch(void* const* d_in, const int* in_sizes, int n_in,
                              void* d_out, int out_size, void* d_ws, size_t ws_size,
                              hipStream_t stream) {
  (void)in_sizes; (void)n_in; (void)out_size; (void)ws_size;
  const float* samples  = (const float*)d_in[0];
  const int*   actions  = (const int*)  d_in[1];
  const float* W_embed  = (const float*)d_in[2];
  const float* b_embed  = (const float*)d_in[3];
  const float* act_emb  = (const float*)d_in[4];
  const float* ln_w     = (const float*)d_in[5];
  const float* ln_b     = (const float*)d_in[6];
  const float* W_in     = (const float*)d_in[7];
  const float* conv_w   = (const float*)d_in[8];
  const float* conv_b   = (const float*)d_in[9];
  const float* W_xproj  = (const float*)d_in[10];
  const float* W_dt     = (const float*)d_in[11];
  const float* b_dt     = (const float*)d_in[12];
  const float* A_log    = (const float*)d_in[13];
  const float* D_skip   = (const float*)d_in[14];
  const float* W_out    = (const float*)d_in[15];
  const float* normf_w  = (const float*)d_in[16];
  const float* normf_b  = (const float*)d_in[17];
  const float* outln_w  = (const float*)d_in[18];
  const float* outln_b  = (const float*)d_in[19];

  char* ws = (char*)d_ws;
  size_t off = 0;
  auto carve = [&](size_t bytes) -> void* {
    void* p = ws + off;
    off += (bytes + 255) & ~(size_t)255;
    return p;
  };
  unsigned short* sbf     = (unsigned short*)carve((size_t)M * STOCH * 2);
  unsigned short* pWemb   = (unsigned short*)carve((size_t)STOCH * DM * 2);
  unsigned short* pWin    = (unsigned short*)carve((size_t)NL * DM * 2 * DI * 2);
  unsigned short* pWxp    = (unsigned short*)carve((size_t)NL * DI * 64 * 2);
  unsigned short* pWdt    = (unsigned short*)carve((size_t)NL * DTR * DI * 2);
  unsigned short* pWout   = (unsigned short*)carve((size_t)NL * DI * DM * 2);
  float*          Aneg    = (float*)carve((size_t)NL * DI * DST * 4);
  float*          x       = (float*)carve((size_t)M * DM * 4);
  unsigned short* xnbf    = (unsigned short*)carve((size_t)M * DM * 2);
  float*          xz      = (float*)carve((size_t)M * 2 * DI * 4);
  float*          xc      = (float*)carve((size_t)M * DI * 4);
  unsigned short* xcbf    = (unsigned short*)carve((size_t)M * DI * 2);
  float*          dbc     = (float*)carve((size_t)M * 64 * 4);
  unsigned short* dtrawbf = (unsigned short*)carve((size_t)M * DTR * 2);
  float*          dtpre   = (float*)carve((size_t)M * DI * 4);
  unsigned short* ybf     = (unsigned short*)carve((size_t)M * DI * 2);
  float*          ln2     = (float*)carve((size_t)M * DM * 4);

  // ---- prep: convert activations, pack weights, A = -exp(A_log) ----
  {
    int n = M * STOCH;
    k_f32_to_bf16<<<(n + 255) / 256, 256, 0, stream>>>(samples, sbf, n);
  }
  k_pack_b<<<(STOCH * DM + 255) / 256, 256, 0, stream>>>(W_embed, pWemb, STOCH, DM);
  for (int l = 0; l < NL; ++l) {
    k_pack_b<<<(DM * 2 * DI + 255) / 256, 256, 0, stream>>>(
        W_in + (size_t)l * DM * 2 * DI, pWin + (size_t)l * DM * 2 * DI, DM, 2 * DI);
    k_pack_b<<<(DI * 64 + 255) / 256, 256, 0, stream>>>(
        W_xproj + (size_t)l * DI * 64, pWxp + (size_t)l * DI * 64, DI, 64);
    k_pack_b<<<(DTR * DI + 255) / 256, 256, 0, stream>>>(
        W_dt + (size_t)l * DTR * DI, pWdt + (size_t)l * DTR * DI, DTR, DI);
    k_pack_b<<<(DI * DM + 255) / 256, 256, 0, stream>>>(
        W_out + (size_t)l * DI * DM, pWout + (size_t)l * DI * DM, DI, DM);
  }
  {
    int n = NL * DI * DST;
    k_aneg<<<(n + 255) / 256, 256, 0, stream>>>(A_log, Aneg, n);
  }

  // ---- embedding: x = samples @ W_embed + b_embed + act_embed[actions] ----
  k_gemm_bf16<<<dim3(DM / 64, M / 64), 128, 0, stream>>>(sbf, pWemb, x, STOCH, DM, 0);
  k_embed_epi<<<(M * DM + 255) / 256, 256, 0, stream>>>(x, b_embed, act_emb, actions, M * DM);

  // ---- mamba layers ----
  for (int l = 0; l < NL; ++l) {
    k_ln_bf16<<<M, 256, 0, stream>>>(x, ln_w + (size_t)l * DM, ln_b + (size_t)l * DM, xnbf);
    k_gemm_bf16<<<dim3((2 * DI) / 64, M / 64), 128, 0, stream>>>(
        xnbf, pWin + (size_t)l * DM * 2 * DI, xz, DM, 2 * DI, 0);
    k_conv_silu<<<dim3(DI / 256, T, Bsz), 256, 0, stream>>>(
        xz, conv_w + (size_t)l * DI * 4, conv_b + (size_t)l * DI, xc, xcbf);
    k_gemm_bf16<<<dim3(64 / 64, M / 64), 128, 0, stream>>>(
        xcbf, pWxp + (size_t)l * DI * 64, dbc, DI, 64, 0);
    k_dtraw<<<(M * DTR + 255) / 256, 256, 0, stream>>>(dbc, dtrawbf, M * DTR);
    k_gemm_bf16<<<dim3(DI / 64, M / 64), 128, 0, stream>>>(
        dtrawbf, pWdt + (size_t)l * DTR * DI, dtpre, DTR, DI, 0);
    k_scan<<<dim3(DI / 256, Bsz), 256, 0, stream>>>(
        xc, dtpre, dbc, xz, Aneg + (size_t)l * DI * DST,
        b_dt + (size_t)l * DI, D_skip + (size_t)l * DI, ybf);
    k_gemm_bf16<<<dim3(DM / 64, M / 64), 128, 0, stream>>>(
        ybf, pWout + (size_t)l * DI * DM, x, DI, DM, 1);   // residual add
  }

  // ---- final double layernorm + mean over T ----
  k_final_ln<<<M, 256, 0, stream>>>(x, normf_w, normf_b, outln_w, outln_b, ln2);
  k_mean<<<dim3(Bsz, DM / 256), 256, 0, stream>>>(ln2, (float*)d_out);
}
